// MemoryEfficientAttention_43825846288473
// MI455X (gfx1250) — matrix operand order repaired and checked by reading, not compiled, not run on hardware
//
#include <hip/hip_runtime.h>

// ---------------------------------------------------------------------------
// MI455X (gfx1250) multi-head attention forward, f16 WMMA with f32 accumulate.
// Compute-bound (~70 GFLOP vs ~6us of HBM traffic at 23.3 TB/s; working set
// fits in 192MB L2). All matmuls run through v_wmma_f32_16x16x32_f16, with
// software-pipelined (double-buffered) fragment loads so WMMAs don't drain
// loadcnt to zero every issue.
// ---------------------------------------------------------------------------

typedef __attribute__((ext_vector_type(16))) _Float16 v16h;
typedef __attribute__((ext_vector_type(8)))  _Float16 v8h;
typedef __attribute__((ext_vector_type(4)))  _Float16 v4h;
typedef __attribute__((ext_vector_type(8)))  float    v8f;

#define D_MODEL 1024
#define N_HEADS 16
#define D_HEAD  64
#define SEQ     2048
#define BATCH   2
#define ROWS    (BATCH * SEQ)     /* 4096 */
#define QK_SCALE 0.125f           /* 64^-0.5 */
#define NCHUNK  (SEQ / 32)        /* 64 key chunks of 32 */

// ---- WMMA wrapper: D = A(16x32 f16) * B(32x16 f16) + C(16x16 f32) ----------
static __device__ __forceinline__ v8f wmma16(v16h a, v16h b, v8f c) {
  return __builtin_amdgcn_wmma_f32_16x16x32_f16(false, a, false, b,
                                                (short)0, c, false, false);
}

// A-fragment (16x32, M x K), row-major source with leading dim `ld` (halves).
// ISA layout: lanes 0-15 row M=lane, K 0..7 in a[0..7], K 16..23 in a[8..15];
//             lanes 16-31 row M=lane-16, K 8..15 and 24..31.
static __device__ __forceinline__ v16h load_A_frag(const _Float16* base, int ld,
                                                   int k0, int lane) {
  const int mr = lane & 15, hg = lane >> 4;
  const _Float16* p = base + (size_t)mr * ld + k0 + hg * 8;
  v8h lo = *(const v8h*)p;
  v8h hi = *(const v8h*)(p + 16);
  v16h a;
#pragma unroll
  for (int i = 0; i < 8; ++i) { a[i] = lo[i]; a[i + 8] = hi[i]; }
  return a;
}

// B-fragment (32x16, K x N) where B[k][n] = base[n][k] (base row-major over k,
// leading dim `ld`). ISA layout: lanes 0-15 hold K 0..7 and 16..23 (col N=lane),
// lanes 16-31 hold K 8..15 and 24..31 (col N=lane-16), matching the A-fragment.
static __device__ __forceinline__ v16h load_B_frag(const _Float16* base, int ld,
                                                   int k0, int lane) {
  const int nr = lane & 15, hg = lane >> 4;
  const _Float16* p = base + (size_t)nr * ld + k0 + hg * 8;
  v8h lo = *(const v8h*)p;
  v8h hi = *(const v8h*)(p + 16);
  v16h b;
#pragma unroll
  for (int i = 0; i < 8; ++i) { b[i] = lo[i]; b[i + 8] = hi[i]; }
  return b;
}

// ---- software-pipelined 16x64 GEMM tile over K = D_MODEL -------------------
// acc[t] += A(16 x 1024) * B_t(1024 x 16), B[k][n] = W[n][k] (torch Linear).
static __device__ __forceinline__ void gemm_16x64(const _Float16* __restrict__ Abase,
                                                  const _Float16* __restrict__ Wbase,
                                                  int lane, v8f (&acc)[4]) {
  v16h a_cur = load_A_frag(Abase, D_MODEL, 0, lane);
  v16h b_cur[4];
#pragma unroll
  for (int t = 0; t < 4; ++t)
    b_cur[t] = load_B_frag(Wbase + (size_t)t * 16 * D_MODEL, D_MODEL, 0, lane);

  for (int k0 = 0; k0 < D_MODEL - 32; k0 += 32) {
    // issue next-iteration fragment loads before consuming current ones
    v16h a_nxt = load_A_frag(Abase, D_MODEL, k0 + 32, lane);
    v16h b_nxt[4];
#pragma unroll
    for (int t = 0; t < 4; ++t)
      b_nxt[t] = load_B_frag(Wbase + (size_t)t * 16 * D_MODEL, D_MODEL, k0 + 32, lane);
#pragma unroll
    for (int t = 0; t < 4; ++t) acc[t] = wmma16(a_cur, b_cur[t], acc[t]);
    a_cur = a_nxt;
#pragma unroll
    for (int t = 0; t < 4; ++t) b_cur[t] = b_nxt[t];
  }
#pragma unroll
  for (int t = 0; t < 4; ++t) acc[t] = wmma16(a_cur, b_cur[t], acc[t]);
}

// ---- f32 -> f16 conversion (x and the four weight matrices) ----------------
__global__ __launch_bounds__(256) void cvt_f32_f16_kernel(
    const float* __restrict__ src, _Float16* __restrict__ dst, int n) {
  const int i = (blockIdx.x * 256 + threadIdx.x) * 4;
  if (i < n) {
    float4 v = *(const float4*)(src + i);
    v4h h;
    h[0] = (_Float16)v.x; h[1] = (_Float16)v.y;
    h[2] = (_Float16)v.z; h[3] = (_Float16)v.w;
    *(v4h*)(dst + i) = h;
  }
}

// ---- fused QKV projection: out[s][e] = sum_d x[s,d] * W[e,d] + b[e] --------
// blockIdx.y selects Q (0), K (1), V (2). One wave per 16x64 output tile.
// Q,K stored f16 [B,H,S,64]; V stored transposed f16 [B,H,64,S].
__global__ __launch_bounds__(256) void qkv_gemm_kernel(
    const _Float16* __restrict__ xh,
    const _Float16* __restrict__ Wqh, const _Float16* __restrict__ Wkh,
    const _Float16* __restrict__ Wvh,
    const float* __restrict__ bq, const float* __restrict__ bk,
    const float* __restrict__ bv,
    _Float16* __restrict__ Qh, _Float16* __restrict__ Kh,
    _Float16* __restrict__ Vth) {
  const int mode = blockIdx.y;
  const _Float16* W = (mode == 0) ? Wqh : ((mode == 1) ? Wkh : Wvh);
  const float* bias = (mode == 0) ? bq : ((mode == 1) ? bk : bv);
  _Float16* Out = (mode == 0) ? Qh : ((mode == 1) ? Kh : Vth);

  const int lane = threadIdx.x & 31;
  const int waveId = blockIdx.x * 8 + (threadIdx.x >> 5);  // 0..4095
  const int mtile = waveId >> 4;   // 0..255 (16 rows each)
  const int n4 = waveId & 15;      // 0..15  (64 cols each)

  v8f acc[4] = {};
  gemm_16x64(xh + (size_t)mtile * 16 * D_MODEL,
             W + (size_t)n4 * 64 * D_MODEL, lane, acc);

  // Epilogue: bias + scatter into per-head layout.
#pragma unroll
  for (int t = 0; t < 4; ++t) {
    const int e = n4 * 64 + t * 16 + (lane & 15);
    const float be = bias[e];
    const int h = e >> 6, dh = e & 63;
#pragma unroll
    for (int r = 0; r < 8; ++r) {
      const int srow = mtile * 16 + r + (lane >> 4) * 8;   // 0..4095
      const int bb = srow >> 11, sq = srow & 2047;
      const _Float16 val = (_Float16)(acc[t][r] + be);
      if (mode != 2)
        Out[(((size_t)(bb * N_HEADS + h)) * SEQ + sq) * D_HEAD + dh] = val;
      else
        Out[(((size_t)(bb * N_HEADS + h)) * D_HEAD + dh) * SEQ + sq] = val;
    }
  }
}

// ---- flash-style attention: one wave per 16-query tile ---------------------
__global__ __launch_bounds__(256) void attn_kernel(
    const _Float16* __restrict__ Qh, const _Float16* __restrict__ Kh,
    const _Float16* __restrict__ Vth, _Float16* __restrict__ Ah) {
  __shared__ __attribute__((aligned(16))) _Float16 Pbuf[8][16 * 32];

  const int lane = threadIdx.x & 31;
  const int wv = threadIdx.x >> 5;
  const int hg = lane >> 4, nr = lane & 15;
  const int bh = blockIdx.y;                 // 0..31 (b*16+h)
  const int qbase = (blockIdx.x * 8 + wv) * 16;

  const _Float16* Qp = Qh + ((size_t)bh * SEQ + qbase) * D_HEAD;
  const _Float16* Kp = Kh + (size_t)bh * SEQ * D_HEAD;
  const _Float16* Vp = Vth + (size_t)bh * D_HEAD * SEQ;

  const v16h qa0 = load_A_frag(Qp, D_HEAD, 0, lane);
  const v16h qa1 = load_A_frag(Qp, D_HEAD, 32, lane);

  float m[8], l[8];
  v8f o[4] = {};
#pragma unroll
  for (int r = 0; r < 8; ++r) { m[r] = -1e30f; l[r] = 0.f; }

  _Float16* Pw = &Pbuf[wv][0];

  // preload K fragments for chunk 0
  v16h kb[2][2];
#pragma unroll
  for (int sub = 0; sub < 2; ++sub) {
    const _Float16* Ksub = Kp + (size_t)(sub * 16) * D_HEAD;
    kb[sub][0] = load_B_frag(Ksub, D_HEAD, 0, lane);
    kb[sub][1] = load_B_frag(Ksub, D_HEAD, 32, lane);
  }

  for (int kc = 0; kc < NCHUNK; ++kc) {
    const int kbase = kc * 32;

    // V fragments for the current chunk: independent of softmax/P, issue early
    v16h vb[4];
#pragma unroll
    for (int t = 0; t < 4; ++t)
      vb[t] = load_B_frag(Vp + (size_t)(t * 16) * SEQ, SEQ, kbase, lane);

    // scores with preloaded K fragments (no load wait needed here)
    v8f st[2];
#pragma unroll
    for (int sub = 0; sub < 2; ++sub) {
      v8f s = {};
      s = wmma16(qa0, kb[sub][0], s);
      s = wmma16(qa1, kb[sub][1], s);
      st[sub] = s;
    }

    // issue next chunk's K fragments; they fly during the softmax VALU section
    if (kc + 1 < NCHUNK) {
      const int kn = kbase + 32;
#pragma unroll
      for (int sub = 0; sub < 2; ++sub) {
        const _Float16* Ksub = Kp + (size_t)(kn + sub * 16) * D_HEAD;
        kb[sub][0] = load_B_frag(Ksub, D_HEAD, 0, lane);
        kb[sub][1] = load_B_frag(Ksub, D_HEAD, 32, lane);
      }
    }

    // online softmax; C/D layout: VGPR r -> row r (lanes<16) / r+8 (lanes>=16),
    // so xor-shuffles over masks 1,2,4,8 reduce exactly one score row.
#pragma unroll
    for (int r = 0; r < 8; ++r) {
      const float v0 = st[0][r] * QK_SCALE;
      const float v1 = st[1][r] * QK_SCALE;
      float mx = fmaxf(v0, v1);
#pragma unroll
      for (int off = 1; off < 16; off <<= 1)
        mx = fmaxf(mx, __shfl_xor(mx, off, 32));
      const float mnew = fmaxf(m[r], mx);
      const float p0 = __expf(v0 - mnew);
      const float p1 = __expf(v1 - mnew);
      float rs = p0 + p1;
#pragma unroll
      for (int off = 1; off < 16; off <<= 1)
        rs += __shfl_xor(rs, off, 32);
      const float alpha = __expf(m[r] - mnew);
      l[r] = l[r] * alpha + rs;
      m[r] = mnew;
#pragma unroll
      for (int t = 0; t < 4; ++t) o[t][r] *= alpha;
      // stage probs (D-layout -> row-major 16x32 in LDS)
      const int prow = r + hg * 8;
      Pw[prow * 32 + nr] = (_Float16)p0;
      Pw[prow * 32 + 16 + nr] = (_Float16)p1;
    }
    asm volatile("" ::: "memory");  // keep LDS store->load order (DS in-order/wave)

    // P(16x32) @ V(32x64): reload P as an A-fragment; vb has been in flight
    // since the top of the iteration.
    const v16h pa = load_A_frag(Pw, 32, 0, lane);
#pragma unroll
    for (int t = 0; t < 4; ++t) o[t] = wmma16(pa, vb[t], o[t]);
    asm volatile("" ::: "memory");
  }

  // normalize and write [B,S,D_MODEL] f16 (pre-output-projection)
  const int bb = bh >> 4, h = bh & 15;
#pragma unroll
  for (int r = 0; r < 8; ++r) {
    const float inv = 1.0f / l[r];
    const int srow = bb * SEQ + qbase + r + hg * 8;
#pragma unroll
    for (int t = 0; t < 4; ++t) {
      const int col = h * D_HEAD + t * 16 + nr;
      Ah[(size_t)srow * D_MODEL + col] = (_Float16)(o[t][r] * inv);
    }
  }
}

// ---- output projection: out[s][e] = sum_d A[s,d] * Wo[e,d] + bo[e] (f32) ---
__global__ __launch_bounds__(256) void out_proj_kernel(
    const _Float16* __restrict__ Ah, const _Float16* __restrict__ Woh,
    const float* __restrict__ bo, float* __restrict__ out) {
  const int lane = threadIdx.x & 31;
  const int waveId = blockIdx.x * 8 + (threadIdx.x >> 5);
  const int mtile = waveId >> 4;
  const int n4 = waveId & 15;

  v8f acc[4] = {};
  gemm_16x64(Ah + (size_t)mtile * 16 * D_MODEL,
             Woh + (size_t)n4 * 64 * D_MODEL, lane, acc);

#pragma unroll
  for (int t = 0; t < 4; ++t) {
    const int e = n4 * 64 + t * 16 + (lane & 15);
    const float be = bo[e];
#pragma unroll
    for (int r = 0; r < 8; ++r) {
      const int srow = mtile * 16 + r + (lane >> 4) * 8;
      out[(size_t)srow * D_MODEL + e] = acc[t][r] + be;
    }
  }
}

// ---------------------------------------------------------------------------
extern "C" void kernel_launch(void* const* d_in, const int* in_sizes, int n_in,
                              void* d_out, int out_size, void* d_ws, size_t ws_size,
                              hipStream_t stream) {
  const float* x  = (const float*)d_in[0];
  const float* Wq = (const float*)d_in[1];
  const float* bq = (const float*)d_in[2];
  const float* Wk = (const float*)d_in[3];
  const float* bk = (const float*)d_in[4];
  const float* Wv = (const float*)d_in[5];
  const float* bv = (const float*)d_in[6];
  const float* Wo = (const float*)d_in[7];
  const float* bo = (const float*)d_in[8];
  float* out = (float*)d_out;

  // workspace carving (all offsets 256B aligned); total ~48 MiB
  char* ws = (char*)d_ws;
  size_t off = 0;
  auto carve = [&](size_t bytes) {
    char* p = ws + off;
    off += (bytes + 255) & ~(size_t)255;
    return p;
  };
  _Float16* xh  = (_Float16*)carve((size_t)ROWS * D_MODEL * 2);
  _Float16* Wqh = (_Float16*)carve((size_t)D_MODEL * D_MODEL * 2);
  _Float16* Wkh = (_Float16*)carve((size_t)D_MODEL * D_MODEL * 2);
  _Float16* Wvh = (_Float16*)carve((size_t)D_MODEL * D_MODEL * 2);
  _Float16* Woh = (_Float16*)carve((size_t)D_MODEL * D_MODEL * 2);
  _Float16* Qh  = (_Float16*)carve((size_t)ROWS * D_MODEL * 2);
  _Float16* Kh  = (_Float16*)carve((size_t)ROWS * D_MODEL * 2);
  _Float16* Vth = (_Float16*)carve((size_t)ROWS * D_MODEL * 2);
  _Float16* Ah  = (_Float16*)carve((size_t)ROWS * D_MODEL * 2);

  const int nx = ROWS * D_MODEL;       // 4,194,304
  const int nw = D_MODEL * D_MODEL;    // 1,048,576
  cvt_f32_f16_kernel<<<nx / 1024, 256, 0, stream>>>(x, xh, nx);
  cvt_f32_f16_kernel<<<nw / 1024, 256, 0, stream>>>(Wq, Wqh, nw);
  cvt_f32_f16_kernel<<<nw / 1024, 256, 0, stream>>>(Wk, Wkh, nw);
  cvt_f32_f16_kernel<<<nw / 1024, 256, 0, stream>>>(Wv, Wvh, nw);
  cvt_f32_f16_kernel<<<nw / 1024, 256, 0, stream>>>(Wo, Woh, nw);

  // 256 mtiles * 16 n4-tiles = 4096 waves = 512 blocks of 8 waves; y: Q/K/V
  qkv_gemm_kernel<<<dim3(512, 3), 256, 0, stream>>>(
      xh, Wqh, Wkh, Wvh, bq, bk, bv, Qh, Kh, Vth);

  // 128 q-tiles per (b,h), 8 waves per block -> grid (16, B*H)
  attn_kernel<<<dim3(16, BATCH * N_HEADS), 256, 0, stream>>>(Qh, Kh, Vth, Ah);

  out_proj_kernel<<<512, 256, 0, stream>>>(Ah, Woh, bo, out);
}
